// GraphTransformerLayer_71751723647463
// MI455X (gfx1250) — compile-verified, matrix operand order checked by reading, not run on hardware
//
#include <hip/hip_runtime.h>
#include <math.h>

// Problem constants (match reference)
#define N_NODES 10000
#define N_EDGES 120000
#define C_IN    256
#define C_OUT   128
#define N_HEAD  8
#define N_GRP   16
#define HC      (N_HEAD * C_OUT)   // 1024

// ---------------------------------------------------------------------------
// Types for WMMA fragments (wave32, CDNA5)
// ---------------------------------------------------------------------------
typedef __bf16        v16bf __attribute__((ext_vector_type(16)));
typedef float         v8f   __attribute__((ext_vector_type(8)));
typedef unsigned int  v4u   __attribute__((ext_vector_type(4)));
typedef unsigned int  v2u   __attribute__((ext_vector_type(2)));

union Frag { v4u q[2]; v16bf v; };   // 32 bytes: 16 bf16 per lane

// ---------------------------------------------------------------------------
// Helpers
// ---------------------------------------------------------------------------
__device__ __forceinline__ unsigned short f2bf(float f) {
  unsigned int u = __float_as_uint(f);
  u += 0x7FFFu + ((u >> 16) & 1u);    // round-to-nearest-even
  return (unsigned short)(u >> 16);
}
__device__ __forceinline__ float bf2f(unsigned int h16) {
  return __uint_as_float((h16 & 0xFFFFu) << 16);
}
// Monotone float->uint key so atomicMax on uint == float max (order-invariant)
__device__ __forceinline__ unsigned int fkey(float f) {
  unsigned int b = __float_as_uint(f);
  return (b & 0x80000000u) ? ~b : (b | 0x80000000u);
}
__device__ __forceinline__ float kdec(unsigned int k) {
  return (k & 0x80000000u) ? __uint_as_float(k & 0x7FFFFFFFu)
                           : __uint_as_float(~k);
}

// ---------------------------------------------------------------------------
// f32 -> bf16 convert
// ---------------------------------------------------------------------------
__global__ void k_cvt_bf16(const float* __restrict__ in,
                           unsigned short* __restrict__ out, int n) {
  int i = blockIdx.x * blockDim.x + threadIdx.x;
  if (i < n) out[i] = f2bf(in[i]);
}

// ---------------------------------------------------------------------------
// C = A(bf16, MxK row-major) @ W^T(bf16, NcolsxK row-major) + bias
// Wave computes a 16x64 tile: 4 WMMA accumulators, A-frag reused 4x per step.
// Per-lane frag loads follow the ISA 16-bit A/B layout:
//   lanes 0-15 : K = k0+0..7  and k0+16..23
//   lanes 16-31: K = k0+8..15 and k0+24..31
// i.e. two contiguous 16-byte loads per fragment per lane.
// ---------------------------------------------------------------------------
template<bool OUT_BF16>
__global__ void k_gemm_xWT_wmma(const unsigned short* __restrict__ A,
                                const unsigned short* __restrict__ W,
                                const float* __restrict__ bias,
                                void* __restrict__ Cout,
                                int M, int Kd, int Ncols) {
  const int lane  = threadIdx.x;            // 0..31 (wave32)
  const int mtile = blockIdx.x * blockDim.y + threadIdx.y;
  if (mtile * 16 >= M) return;              // whole wave exits (EXEC stays uniform)
  const int n0   = blockIdx.y * 64;
  const int l15  = lane & 15;
  const int koff = (lane >> 4) << 3;        // 0 or 8

  v8f acc[4] = {};
  const unsigned short* arow = A + (size_t)(mtile * 16 + l15) * Kd;

  for (int k0 = 0; k0 < Kd; k0 += 32) {
    Frag a;
    const v4u* pa = (const v4u*)(arow + k0 + koff);
    a.q[0] = pa[0];                         // K = koff .. koff+7
    a.q[1] = pa[2];                         // K = koff+16 .. koff+23
#pragma unroll
    for (int t = 0; t < 4; ++t) {
      const unsigned short* wrow = W + (size_t)(n0 + t * 16 + l15) * Kd;
      Frag b;
      const v4u* pb = (const v4u*)(wrow + k0 + koff);
      b.q[0] = pb[0];
      b.q[1] = pb[2];
      acc[t] = __builtin_amdgcn_wmma_f32_16x16x32_bf16(
          /*neg_a=*/false, a.v, /*neg_b=*/false, b.v,
          /*c_mod=*/(short)0, acc[t], /*reuse_a=*/false, /*reuse_b=*/false);
    }
  }

  // C/D layout: VGPR r -> M = r (lanes 0-15) or 8+r (lanes 16-31); N = lane&15
  const int rowbase = mtile * 16 + ((lane >> 4) << 3);
#pragma unroll
  for (int t = 0; t < 4; ++t) {
    const int col = n0 + t * 16 + l15;
    const float bb = bias[col];
#pragma unroll
    for (int r = 0; r < 8; ++r) {
      float vv = acc[t][r] + bb;
      size_t idx = (size_t)(rowbase + r) * Ncols + col;
      if (OUT_BF16) ((unsigned short*)Cout)[idx] = f2bf(vv);
      else          ((float*)Cout)[idx]          = vv;
    }
  }
}

// ---------------------------------------------------------------------------
// Per-(edge,head) attention score: warp computes dot(q[dst,h,:], k[src,h,:])
// Q/K are bf16 [N, 1024]; each lane loads 4 bf16 (8 B, coalesced) per side.
// ---------------------------------------------------------------------------
__global__ void k_edge_scores(const unsigned short* __restrict__ Q,
                              const unsigned short* __restrict__ K,
                              const int* __restrict__ ei,
                              float* __restrict__ score,
                              unsigned int* __restrict__ smax) {
  int gid  = blockIdx.x * blockDim.x + threadIdx.x;
  int warp = gid >> 5, lane = gid & 31;
  if (warp >= N_EDGES * N_HEAD) return;
  int e = warp >> 3, h = warp & 7;
  int srcn = ei[e], dstn = ei[N_EDGES + e];
  const unsigned short* qp = Q + (size_t)dstn * HC + h * C_OUT + lane * 4;
  const unsigned short* kp = K + (size_t)srcn * HC + h * C_OUT + lane * 4;
  v2u qu = *(const v2u*)qp;
  v2u ku = *(const v2u*)kp;
  float s = bf2f(qu.x) * bf2f(ku.x) + bf2f(qu.x >> 16) * bf2f(ku.x >> 16) +
            bf2f(qu.y) * bf2f(ku.y) + bf2f(qu.y >> 16) * bf2f(ku.y >> 16);
#pragma unroll
  for (int m = 16; m > 0; m >>= 1) s += __shfl_xor(s, m, 32);
  if (lane == 0) {
    s *= 0.088388347648318447f;   // 1/sqrt(128)
    score[warp] = s;
    atomicMax(smax + dstn * N_HEAD + h, fkey(s));
  }
}

// ---------------------------------------------------------------------------
// alpha = exp(score - segmax[dst]);  denom[dst,h] += alpha
// ---------------------------------------------------------------------------
__global__ void k_edge_exp(const float* __restrict__ score,
                           const unsigned int* __restrict__ smax,
                           const int* __restrict__ ei,
                           float* __restrict__ alpha,
                           float* __restrict__ den) {
  int i = blockIdx.x * blockDim.x + threadIdx.x;
  if (i >= N_EDGES * N_HEAD) return;
  int e = i >> 3, h = i & 7;
  int dstn = ei[N_EDGES + e];
  float m = kdec(smax[dstn * N_HEAD + h]);
  float a = __expf(score[i] - m);
  alpha[i] = a;
  atomicAdd(den + dstn * N_HEAD + h, a);
}

// ---------------------------------------------------------------------------
// Warp per edge: out_acc[dst, :] += sum_h (alpha/denom) * v[src,h,:] / H
// ---------------------------------------------------------------------------
__global__ void k_edge_scatter(const unsigned short* __restrict__ V,
                               const float* __restrict__ alpha,
                               const float* __restrict__ den,
                               const int* __restrict__ ei,
                               float* __restrict__ oacc) {
  int gid  = blockIdx.x * blockDim.x + threadIdx.x;
  int e = gid >> 5, lane = gid & 31;
  if (e >= N_EDGES) return;
  int srcn = ei[e], dstn = ei[N_EDGES + e];
  int c = lane * 4;
  float a0 = 0.f, a1 = 0.f, a2 = 0.f, a3 = 0.f;
#pragma unroll
  for (int h = 0; h < N_HEAD; ++h) {
    float w = alpha[(size_t)e * N_HEAD + h] / den[dstn * N_HEAD + h];
    const unsigned short* vp = V + (size_t)srcn * HC + h * C_OUT + c;
    v2u vu = *(const v2u*)vp;
    a0 += w * bf2f(vu.x);       a1 += w * bf2f(vu.x >> 16);
    a2 += w * bf2f(vu.y);       a3 += w * bf2f(vu.y >> 16);
  }
  float* op = oacc + (size_t)dstn * C_OUT + c;
  atomicAdd(op + 0, a0 * 0.125f);   // fold head-mean (1/H)
  atomicAdd(op + 1, a1 * 0.125f);
  atomicAdd(op + 2, a2 * 0.125f);
  atomicAdd(op + 3, a3 * 0.125f);
}

// ---------------------------------------------------------------------------
// Warp per node: beta-gated skip + residual; accumulate GraphNorm group sums
// ---------------------------------------------------------------------------
__global__ void k_node_gate(const float* __restrict__ oacc,
                            const float* __restrict__ xr,
                            const float* __restrict__ xres,
                            const float* __restrict__ Wbeta,
                            const int* __restrict__ batch,
                            float* __restrict__ hbuf,
                            float* __restrict__ gsum,
                            float* __restrict__ gcnt) {
  int gid = blockIdx.x * blockDim.x + threadIdx.x;
  int n = gid >> 5, lane = gid & 31;
  if (n >= N_NODES) return;
  int c = lane * 4;
  float o[4], r[4];
  float part = 0.f;
#pragma unroll
  for (int j = 0; j < 4; ++j) {
    o[j] = oacc[(size_t)n * C_OUT + c + j];
    r[j] = xr[(size_t)n * C_OUT + c + j];
    part += o[j] * Wbeta[c + j] + r[j] * Wbeta[C_OUT + c + j] +
            (o[j] - r[j]) * Wbeta[2 * C_OUT + c + j];
  }
#pragma unroll
  for (int m = 16; m > 0; m >>= 1) part += __shfl_xor(part, m, 32);
  float beta = 1.f / (1.f + __expf(-part));
  int g = batch[n];
#pragma unroll
  for (int j = 0; j < 4; ++j) {
    float hv = beta * r[j] + (1.f - beta) * o[j] + xres[(size_t)n * C_OUT + c + j];
    hbuf[(size_t)n * C_OUT + c + j] = hv;
    atomicAdd(gsum + g * C_OUT + c + j, hv);
  }
  if (lane == 0) atomicAdd(gcnt + g, 1.0f);
}

// mean = gsum / max(cnt,1)
__global__ void k_group_mean(const float* __restrict__ gsum,
                             const float* __restrict__ gcnt,
                             float* __restrict__ mean) {
  int i = blockIdx.x * blockDim.x + threadIdx.x;
  if (i >= N_GRP * C_OUT) return;
  float cnt = fmaxf(gcnt[i >> 7], 1.f);
  mean[i] = gsum[i] / cnt;
}

// accumulate centered^2 per group
__global__ void k_var_acc(const float* __restrict__ hbuf,
                          const float* __restrict__ mean,
                          const float* __restrict__ gms,
                          const int* __restrict__ batch,
                          float* __restrict__ vsum) {
  int i = blockIdx.x * blockDim.x + threadIdx.x;
  if (i >= N_NODES * C_OUT) return;
  int n = i >> 7, c = i & 127;
  int g = batch[n];
  float ctr = hbuf[i] - gms[c] * mean[g * C_OUT + c];
  atomicAdd(vsum + g * C_OUT + c, ctr * ctr);
}

__global__ void k_group_var(const float* __restrict__ vsum,
                            const float* __restrict__ gcnt,
                            float* __restrict__ var) {
  int i = blockIdx.x * blockDim.x + threadIdx.x;
  if (i >= N_GRP * C_OUT) return;
  float cnt = fmaxf(gcnt[i >> 7], 1.f);
  var[i] = vsum[i] / cnt;
}

// GraphNorm affine + exact-erf GELU
__global__ void k_norm_gelu(const float* __restrict__ hbuf,
                            const float* __restrict__ mean,
                            const float* __restrict__ var,
                            const float* __restrict__ gms,
                            const float* __restrict__ gnw,
                            const float* __restrict__ gnb,
                            const int* __restrict__ batch,
                            float* __restrict__ out) {
  int i = blockIdx.x * blockDim.x + threadIdx.x;
  if (i >= N_NODES * C_OUT) return;
  int n = i >> 7, c = i & 127;
  int g = batch[n];
  float ctr = hbuf[i] - gms[c] * mean[g * C_OUT + c];
  float y = gnw[c] * ctr * rsqrtf(var[g * C_OUT + c] + 1e-5f) + gnb[c];
  out[i] = 0.5f * y * (1.f + erff(y * 0.70710678118654752f));
}

// ---------------------------------------------------------------------------
// Host launcher
// ---------------------------------------------------------------------------
static inline size_t alignup(size_t v) { return (v + 255) & ~(size_t)255; }

extern "C" void kernel_launch(void* const* d_in, const int* in_sizes, int n_in,
                              void* d_out, int out_size, void* d_ws, size_t ws_size,
                              hipStream_t stream) {
  const float* x      = (const float*)d_in[0];
  const int*   ei     = (const int*)d_in[1];
  const int*   batch  = (const int*)d_in[2];
  const float* Wq     = (const float*)d_in[3];
  const float* bq     = (const float*)d_in[4];
  const float* Wk     = (const float*)d_in[5];
  const float* bk     = (const float*)d_in[6];
  const float* Wv     = (const float*)d_in[7];
  const float* bv     = (const float*)d_in[8];
  const float* Wskip  = (const float*)d_in[9];
  const float* bskip  = (const float*)d_in[10];
  const float* Wbeta  = (const float*)d_in[11];
  const float* Wres   = (const float*)d_in[12];
  const float* bres   = (const float*)d_in[13];
  const float* gnw    = (const float*)d_in[14];
  const float* gnb    = (const float*)d_in[15];
  const float* gms    = (const float*)d_in[16];
  float* out = (float*)d_out;

  // Workspace carve-up
  char* w = (char*)d_ws;
  size_t off = 0;
  auto take = [&](size_t bytes) { char* p = w + off; off += alignup(bytes); return p; };
  unsigned short* xbf   = (unsigned short*)take((size_t)N_NODES * C_IN * 2);
  unsigned short* wqbf  = (unsigned short*)take((size_t)HC * C_IN * 2);
  unsigned short* wkbf  = (unsigned short*)take((size_t)HC * C_IN * 2);
  unsigned short* wvbf  = (unsigned short*)take((size_t)HC * C_IN * 2);
  unsigned short* wskbf = (unsigned short*)take((size_t)C_OUT * C_IN * 2);
  unsigned short* wrsbf = (unsigned short*)take((size_t)C_OUT * C_IN * 2);
  unsigned short* Qb    = (unsigned short*)take((size_t)N_NODES * HC * 2);
  unsigned short* Kb    = (unsigned short*)take((size_t)N_NODES * HC * 2);
  unsigned short* Vb    = (unsigned short*)take((size_t)N_NODES * HC * 2);
  float* xr   = (float*)take((size_t)N_NODES * C_OUT * 4);
  float* xres = (float*)take((size_t)N_NODES * C_OUT * 4);
  float* score = (float*)take((size_t)N_EDGES * N_HEAD * 4);
  float* alpha = (float*)take((size_t)N_EDGES * N_HEAD * 4);
  unsigned int* smax = (unsigned int*)take((size_t)N_NODES * N_HEAD * 4);
  float* den  = (float*)take((size_t)N_NODES * N_HEAD * 4);
  float* oacc = (float*)take((size_t)N_NODES * C_OUT * 4);
  float* hbuf = (float*)take((size_t)N_NODES * C_OUT * 4);
  float* gsum = (float*)take((size_t)N_GRP * C_OUT * 4);
  float* gcnt = (float*)take((size_t)N_GRP * 4);
  float* vsum = (float*)take((size_t)N_GRP * C_OUT * 4);
  float* mean = (float*)take((size_t)N_GRP * C_OUT * 4);
  float* var  = (float*)take((size_t)N_GRP * C_OUT * 4);

  // Zero the accumulators (graph-capture-legal async memsets)
  hipMemsetAsync(smax, 0, (size_t)N_NODES * N_HEAD * 4, stream);
  hipMemsetAsync(den,  0, (size_t)N_NODES * N_HEAD * 4, stream);
  hipMemsetAsync(oacc, 0, (size_t)N_NODES * C_OUT * 4, stream);
  hipMemsetAsync(gsum, 0, (size_t)N_GRP * C_OUT * 4, stream);
  hipMemsetAsync(gcnt, 0, (size_t)N_GRP * 4, stream);
  hipMemsetAsync(vsum, 0, (size_t)N_GRP * C_OUT * 4, stream);

  // bf16 conversions
  auto cvt = [&](const float* src, unsigned short* dst, int n) {
    k_cvt_bf16<<<(n + 255) / 256, 256, 0, stream>>>(src, dst, n);
  };
  cvt(x,     xbf,   N_NODES * C_IN);
  cvt(Wq,    wqbf,  HC * C_IN);
  cvt(Wk,    wkbf,  HC * C_IN);
  cvt(Wv,    wvbf,  HC * C_IN);
  cvt(Wskip, wskbf, C_OUT * C_IN);
  cvt(Wres,  wrsbf, C_OUT * C_IN);

  // WMMA GEMMs: Q/K/V (Ncols=1024, bf16 out), skip/res (Ncols=128, f32 out)
  dim3 gb(32, 4);
  dim3 gridQKV((N_NODES / 16 + 3) / 4, HC / 64);
  k_gemm_xWT_wmma<true><<<gridQKV, gb, 0, stream>>>(xbf, wqbf, bq, Qb, N_NODES, C_IN, HC);
  k_gemm_xWT_wmma<true><<<gridQKV, gb, 0, stream>>>(xbf, wkbf, bk, Kb, N_NODES, C_IN, HC);
  k_gemm_xWT_wmma<true><<<gridQKV, gb, 0, stream>>>(xbf, wvbf, bv, Vb, N_NODES, C_IN, HC);
  dim3 gridSR((N_NODES / 16 + 3) / 4, C_OUT / 64);
  k_gemm_xWT_wmma<false><<<gridSR, gb, 0, stream>>>(xbf, wskbf, bskip, xr,   N_NODES, C_IN, C_OUT);
  k_gemm_xWT_wmma<false><<<gridSR, gb, 0, stream>>>(xbf, wrsbf, bres,  xres, N_NODES, C_IN, C_OUT);

  // Attention
  {
    long long th = (long long)N_EDGES * N_HEAD * 32;
    k_edge_scores<<<(unsigned)((th + 255) / 256), 256, 0, stream>>>(Qb, Kb, ei, score, smax);
  }
  k_edge_exp<<<(N_EDGES * N_HEAD + 255) / 256, 256, 0, stream>>>(score, smax, ei, alpha, den);
  {
    long long th = (long long)N_EDGES * 32;
    k_edge_scatter<<<(unsigned)((th + 255) / 256), 256, 0, stream>>>(Vb, alpha, den, ei, oacc);
  }

  // Node epilogue + GraphNorm + GELU
  k_node_gate<<<(N_NODES * 32 + 255) / 256, 256, 0, stream>>>(oacc, xr, xres, Wbeta,
                                                              batch, hbuf, gsum, gcnt);
  k_group_mean<<<(N_GRP * C_OUT + 255) / 256, 256, 0, stream>>>(gsum, gcnt, mean);
  k_var_acc<<<(N_NODES * C_OUT + 255) / 256, 256, 0, stream>>>(hbuf, mean, gms, batch, vsum);
  k_group_var<<<(N_GRP * C_OUT + 255) / 256, 256, 0, stream>>>(vsum, gcnt, var);
  k_norm_gelu<<<(N_NODES * C_OUT + 255) / 256, 256, 0, stream>>>(hbuf, mean, var, gms,
                                                                 gnw, gnb, batch, out);
}
